// AdaptiveTopKMoE_81527069212683
// MI455X (gfx1250) — matrix operand order repaired and checked by reading, not compiled, not run on hardware
//
#include <hip/hip_runtime.h>
#include <hip/hip_bf16.h>
#include <math.h>

// Problem constants (B, H, E, K) = (4096, 1024, 16, 2)
#define B_TOK 4096
#define H_DIM 1024
#define E_NUM 16
#define F2    2048   // 2*H, fc1 output width per expert

typedef __attribute__((ext_vector_type(16))) __bf16 bf16x16;
typedef __attribute__((ext_vector_type(8)))  __bf16 bf16x8;
typedef __attribute__((ext_vector_type(8)))  float  f32x8;
typedef __attribute__((ext_vector_type(4)))  float  f32x4;

// ---------------------------------------------------------------------------
// Fragment builders per CDNA5 WMMA VGPR layouts (CDNA5 ISA §7.12.2)
// A (16x32 bf16, MxK): lane L -> row M=L%16, kb=(L>>4)*8, holds K=kb..kb+7
//                      and K=kb+16..kb+23.
// B (32x16 bf16, KxN): lane L -> col N=L%16, kb2=(L>>4)*8, holds K=kb2..kb2+7
//                      and K=kb2+16..kb2+23.
// C (16x16 f32):       vgpr i -> M = i + 8*(L>>4), N = L%16.
// All operands are pre-converted bf16, so fragments are direct vector loads.
// ---------------------------------------------------------------------------

__device__ __forceinline__ bf16x16 a_frag_bf16(const __bf16* __restrict__ p, int kb) {
  bf16x8 lo = *(const bf16x8*)(p + kb);        // K = kb .. kb+7
  bf16x8 hi = *(const bf16x8*)(p + kb + 16);   // K = kb+16 .. kb+23
  return __builtin_shufflevector(lo, hi, 0, 1, 2, 3, 4, 5, 6, 7,
                                 8, 9, 10, 11, 12, 13, 14, 15);
}

__device__ __forceinline__ bf16x16 b_frag_bf16(const __bf16* __restrict__ p, int kb2) {
  bf16x8 lo = *(const bf16x8*)(p + kb2);       // K = kb2 .. kb2+7
  bf16x8 hi = *(const bf16x8*)(p + kb2 + 16);  // K = kb2+16 .. kb2+23
  return __builtin_shufflevector(lo, hi, 0, 1, 2, 3, 4, 5, 6, 7,
                                 8, 9, 10, 11, 12, 13, 14, 15);
}

__device__ __forceinline__ f32x8 wmma_bf16(bf16x16 a, bf16x16 b, f32x8 c) {
  // 8 args: (neg_a, A, neg_b, B, c_mod, C, reuse_a, reuse_b)
  return __builtin_amdgcn_wmma_f32_16x16x32_bf16(false, a, false, b, (short)0, c,
                                                 false, false);
}

// ---------------------------------------------------------------------------
// Kernel 0: zero the per-expert counters
// ---------------------------------------------------------------------------
__global__ void moe_init_kernel(int* __restrict__ cnt) {
  if (threadIdx.x < E_NUM) cnt[threadIdx.x] = 0;
}

// ---------------------------------------------------------------------------
// Kernel 0b: vectorized f32 -> bf16 conversion (8 elements / thread)
// ---------------------------------------------------------------------------
__global__ void moe_cvt_bf16_kernel(const float* __restrict__ src,
                                    __bf16* __restrict__ dst, int n8) {
  const int i = blockIdx.x * blockDim.x + threadIdx.x;
  if (i >= n8) return;
  const f32x4 a = *(const f32x4*)(src + (size_t)i * 8);
  const f32x4 b = *(const f32x4*)(src + (size_t)i * 8 + 4);
  bf16x8 z;
#pragma unroll
  for (int j = 0; j < 4; ++j) {
    z[j]     = (__bf16)a[j];
    z[4 + j] = (__bf16)b[j];
  }
  *(bf16x8*)(dst + (size_t)i * 8) = z;
}

// ---------------------------------------------------------------------------
// Kernel 1: router — one wave32 per token. logits -> top-2 -> softmax ->
//           gates[b*2+k], and append slot id (b*2+k) to expert's token list.
// ---------------------------------------------------------------------------
__global__ void moe_router_kernel(const float* __restrict__ tokens,
                                  const float* __restrict__ router_w,
                                  const float* __restrict__ router_b,
                                  const float* __restrict__ gate_bias,
                                  float* __restrict__ gates,
                                  int* __restrict__ cnt,
                                  int* __restrict__ list) {
  const int gid  = blockIdx.x * blockDim.x + threadIdx.x;
  const int b    = gid >> 5;
  const int lane = gid & 31;
  if (b >= B_TOK) return;

  float acc[E_NUM];
#pragma unroll
  for (int e = 0; e < E_NUM; ++e) acc[e] = 0.0f;

  const float* trow = tokens + (size_t)b * H_DIM;
  for (int h = lane; h < H_DIM; h += 32) {
    const float t = trow[h];
#pragma unroll
    for (int e = 0; e < E_NUM; ++e) acc[e] += t * router_w[e * H_DIM + h];
  }
#pragma unroll
  for (int e = 0; e < E_NUM; ++e) {
#pragma unroll
    for (int off = 16; off > 0; off >>= 1) acc[e] += __shfl_xor(acc[e], off, 32);
  }

  if (lane == 0) {
    float logit[E_NUM];
#pragma unroll
    for (int e = 0; e < E_NUM; ++e) logit[e] = acc[e] + router_b[e] + gate_bias[e];
    // top-1 (strict > keeps lowest index on ties, matching jax.lax.top_k)
    int e0 = 0;
#pragma unroll
    for (int e = 1; e < E_NUM; ++e) if (logit[e] > logit[e0]) e0 = e;
    // top-2
    int e1 = (e0 == 0) ? 1 : 0;
#pragma unroll
    for (int e = 0; e < E_NUM; ++e)
      if (e != e0 && logit[e] > logit[e1]) e1 = e;

    const float s0 = logit[e0], s1 = logit[e1];
    const float m  = fmaxf(s0, s1);
    const float p0 = __expf(s0 - m), p1 = __expf(s1 - m);
    const float inv = 1.0f / (p0 + p1);
    gates[b * 2 + 0] = p0 * inv;
    gates[b * 2 + 1] = p1 * inv;

    const int pos0 = atomicAdd(&cnt[e0], 1);
    list[e0 * B_TOK + pos0] = b * 2 + 0;
    const int pos1 = atomicAdd(&cnt[e1], 1);
    list[e1 * B_TOK + pos1] = b * 2 + 1;
  }
}

// ---------------------------------------------------------------------------
// Kernel 2: fc1 gathered GEMM + exact gelu + gate scale -> gbuf[slot, 0:2048]
// Wave register tile: M=32 x N=64 (2 A-frags x 4 B-frags, 8 WMMA per K-step).
// Block = 256 threads = 8 waves laid out 4(m) x 2(n) -> block tile 128x128.
// B-fragment rows addressed off a single base with compile-time offsets.
// ---------------------------------------------------------------------------
__global__ void moe_fc1_kernel(const __bf16* __restrict__ tokens_bf,
                               const __bf16* __restrict__ w1_bf,
                               const float* __restrict__ gates,
                               const int* __restrict__ cnt,
                               const int* __restrict__ list,
                               __bf16* __restrict__ gbuf) {
  const int e     = blockIdx.z;
  const int count = cnt[e];
  if (blockIdx.y * 128 >= count) return;

  const int lane  = threadIdx.x & 31;
  const int wave  = threadIdx.x >> 5;
  const int m0    = blockIdx.y * 128 + (wave >> 1) * 32;  // wave M base (32 rows)
  const int n0    = blockIdx.x * 128 + (wave & 1) * 64;   // wave N base (64 cols)
  if (m0 >= count) return;

  const int row  = lane & 15;
  const int kb   = (lane >> 4) << 3;   // A fragment K base (8+8 split)
  const int kb2  = (lane >> 4) << 3;   // B fragment K base (8+8 split)

  // Gather the two token rows this lane contributes to (clamp past count).
  const __bf16* arow0;
  const __bf16* arow1;
  {
    const int mi0 = m0 + row;
    const int mi1 = m0 + 16 + row;
    const int s0  = list[e * B_TOK + ((mi0 < count) ? mi0 : m0)];
    const int s1  = list[e * B_TOK + ((mi1 < count) ? mi1 : m0)];
    arow0 = tokens_bf + (size_t)(s0 >> 1) * H_DIM;
    arow1 = tokens_bf + (size_t)(s1 >> 1) * H_DIM;
  }
  // Single B base; ns sub-tiles live at constant offsets ns*16*H_DIM (32KB).
  const __bf16* bbase = w1_bf + (size_t)(e * F2 + n0 + row) * H_DIM;

  f32x8 c[2][4] = {};
  for (int k0 = 0; k0 < H_DIM; k0 += 32) {
    __builtin_prefetch(bbase + k0 + 1024, 0, 1);  // global_prefetch on w1 stream
    bf16x16 a0 = a_frag_bf16(arow0 + k0, kb);
    bf16x16 a1 = a_frag_bf16(arow1 + k0, kb);
    bf16x16 b[4];
#pragma unroll
    for (int ns = 0; ns < 4; ++ns)
      b[ns] = b_frag_bf16(bbase + ns * 16 * H_DIM + k0, kb2);
#pragma unroll
    for (int ns = 0; ns < 4; ++ns) {
      c[0][ns] = wmma_bf16(a0, b[ns], c[0][ns]);
      c[1][ns] = wmma_bf16(a1, b[ns], c[1][ns]);
    }
  }

  // Epilogue: exact gelu, gate scale, bf16 store into per-slot buffer
  const int rbase = (lane >> 4) << 3;
  const int col   = lane & 15;
#pragma unroll
  for (int ms = 0; ms < 2; ++ms) {
#pragma unroll
    for (int i = 0; i < 8; ++i) {
      const int m = m0 + ms * 16 + rbase + i;
      if (m < count) {
        const int s   = list[e * B_TOK + m];
        const float g = gates[s];
#pragma unroll
        for (int ns = 0; ns < 4; ++ns) {
          const float x = c[ms][ns][i];
          const float y = 0.5f * x * (1.0f + erff(x * 0.70710678118654752f)) * g;
          gbuf[(size_t)s * F2 + n0 + ns * 16 + col] = (__bf16)y;
        }
      }
    }
  }
}

// ---------------------------------------------------------------------------
// Kernel 3: gsum[b, f] = gbuf[2b, f] + gbuf[2b+1, f]   (bf16, vectorized x8)
// ---------------------------------------------------------------------------
__global__ void moe_sumslots_kernel(const __bf16* __restrict__ gbuf,
                                    __bf16* __restrict__ gsum) {
  const int i = blockIdx.x * blockDim.x + threadIdx.x;  // over B_TOK*F2/8
  const int per_row = F2 / 8;
  const int b  = i / per_row;
  const int f8 = (i - b * per_row) * 8;
  const bf16x8 x = *(const bf16x8*)(gbuf + (size_t)(2 * b) * F2 + f8);
  const bf16x8 y = *(const bf16x8*)(gbuf + (size_t)(2 * b + 1) * F2 + f8);
  bf16x8 z;
#pragma unroll
  for (int j = 0; j < 8; ++j) z[j] = (__bf16)((float)x[j] + (float)y[j]);
  *(bf16x8*)(gsum + (size_t)b * F2 + f8) = z;
}

// ---------------------------------------------------------------------------
// Kernel 4: fc2 dense GEMM  out(4096x1024) = gsum(4096x2048) @ w2_bf^T
// Same 32x64 wave tile / 128x128 block tile; all-bf16 operands, single bases
// with compile-time sub-tile offsets (64KB steps).
// ---------------------------------------------------------------------------
__global__ void moe_fc2_kernel(const __bf16* __restrict__ gsum,
                               const __bf16* __restrict__ w2_bf,
                               float* __restrict__ out) {
  const int lane = threadIdx.x & 31;
  const int wave = threadIdx.x >> 5;
  const int m0   = blockIdx.y * 128 + (wave >> 1) * 32;  // token rows
  const int n0   = blockIdx.x * 128 + (wave & 1) * 64;   // output features h
  const int row  = lane & 15;
  const int kb   = (lane >> 4) << 3;
  const int kb2  = (lane >> 4) << 3;

  const __bf16* abase = gsum + (size_t)(m0 + row) * F2;   // ms offset = ms*16*F2
  const __bf16* bbase = w2_bf + (size_t)(n0 + row) * F2;  // ns offset = ns*16*F2

  f32x8 c[2][4] = {};
  for (int k0 = 0; k0 < F2; k0 += 32) {
    __builtin_prefetch(bbase + k0 + 1024, 0, 1);
    bf16x16 a0 = a_frag_bf16(abase + k0, kb);
    bf16x16 a1 = a_frag_bf16(abase + 16 * F2 + k0, kb);
    bf16x16 b[4];
#pragma unroll
    for (int ns = 0; ns < 4; ++ns)
      b[ns] = b_frag_bf16(bbase + ns * 16 * F2 + k0, kb2);
#pragma unroll
    for (int ns = 0; ns < 4; ++ns) {
      c[0][ns] = wmma_bf16(a0, b[ns], c[0][ns]);
      c[1][ns] = wmma_bf16(a1, b[ns], c[1][ns]);
    }
  }

  const int rbase = (lane >> 4) << 3;
  const int col   = lane & 15;
#pragma unroll
  for (int ms = 0; ms < 2; ++ms)
#pragma unroll
    for (int i = 0; i < 8; ++i)
#pragma unroll
      for (int ns = 0; ns < 4; ++ns)
        out[(size_t)(m0 + ms * 16 + rbase + i) * H_DIM + n0 + ns * 16 + col] =
            c[ms][ns][i];
}

// ---------------------------------------------------------------------------
// Launch
// ---------------------------------------------------------------------------
extern "C" void kernel_launch(void* const* d_in, const int* in_sizes, int n_in,
                              void* d_out, int out_size, void* d_ws, size_t ws_size,
                              hipStream_t stream) {
  const float* tokens    = (const float*)d_in[0];
  const float* router_w  = (const float*)d_in[1];
  const float* router_b  = (const float*)d_in[2];
  const float* w1        = (const float*)d_in[3];
  const float* w2        = (const float*)d_in[4];
  const float* gate_bias = (const float*)d_in[5];
  float* out = (float*)d_out;

  // Workspace layout (gsum aliases w1_bf: fc1 finishes with w1_bf before
  // sumslots writes gsum; stream order guarantees safety; all regions are
  // fully rewritten every launch -> deterministic under graph replay).
  char* ws = (char*)d_ws;
  float*  gates     = (float*)(ws);                       // 32 KB
  int*    cnt       = (int*)(ws + 32768);                 // 64 B
  int*    list      = (int*)(ws + 33024);                 // 256 KB
  __bf16* tokens_bf = (__bf16*)(ws + (1u << 20));         // @1MB,  8 MB
  __bf16* w2_bf     = (__bf16*)(ws + (9u << 20));         // @9MB,  4 MB
  __bf16* gbuf      = (__bf16*)(ws + (13u << 20));        // @13MB, 32 MB
  __bf16* w1_bf     = (__bf16*)(ws + (45u << 20));        // @45MB, 64 MB
  __bf16* gsum      = (__bf16*)(ws + (45u << 20));        // aliases w1_bf, 16 MB

  moe_init_kernel<<<1, 32, 0, stream>>>(cnt);

  // bf16 pre-conversions (w2: only rows [0,H) are ever used -> first H*F2 elems)
  moe_cvt_bf16_kernel<<<(B_TOK * H_DIM / 8) / 256, 256, 0, stream>>>(
      tokens, tokens_bf, B_TOK * H_DIM / 8);
  moe_cvt_bf16_kernel<<<(E_NUM * F2 * H_DIM / 8) / 256, 256, 0, stream>>>(
      w1, w1_bf, E_NUM * F2 * H_DIM / 8);
  moe_cvt_bf16_kernel<<<(H_DIM * F2 / 8) / 256, 256, 0, stream>>>(
      w2, w2_bf, H_DIM * F2 / 8);

  // one wave32 per token
  moe_router_kernel<<<(B_TOK * 32) / 256, 256, 0, stream>>>(
      tokens, router_w, router_b, gate_bias, gates, cnt, list);

  // gathered fc1 + gelu + gate   (block tile 128x128)
  moe_fc1_kernel<<<dim3(F2 / 128, B_TOK / 128, E_NUM), 256, 0, stream>>>(
      tokens_bf, w1_bf, gates, cnt, list, gbuf);

  // slot-pair reduction (reads gbuf, writes gsum which reuses w1_bf space)
  moe_sumslots_kernel<<<(B_TOK * F2 / 8) / 256, 256, 0, stream>>>(gbuf, gsum);

  // dense fc2   (block tile 128x128)
  moe_fc2_kernel<<<dim3(H_DIM / 128, B_TOK / 128), 256, 0, stream>>>(gsum, w2_bf, out);
}
